// MultiHeadAttention_42460046688339
// MI455X (gfx1250) — matrix operand order repaired and checked by reading, not compiled, not run on hardware
//
#include <hip/hip_runtime.h>

typedef __attribute__((ext_vector_type(16))) _Float16 v16h;
typedef __attribute__((ext_vector_type(8)))  float    v8f;

#define B_  16
#define T_  256
#define D_  2048
#define H_  32
#define HD_ 64
#define SK_ 512
#define M_  (B_*T_)   // 4096

// ---- workspace layout (bytes) ----
#define WS_XA   ((size_t)0)                          // x as f16            [4096 x 2048]
#define WS_W16  (WS_XA  + (size_t)M_*D_*2)           // 4 weights f16 (row-major, untransposed)
#define WS_QROT (WS_W16 + (size_t)4*D_*D_*2)         // rope(q)*0.125 f16   [B,H,T,HD]
#define WS_KVIS (WS_QROT+ (size_t)B_*H_*T_*HD_*2)    // visible K cache f16 [B,H,SK,HD]
#define WS_VT   (WS_KVIS+ (size_t)B_*H_*SK_*HD_*2)   // V cache f16 transposed [B,H,HD,SK]
#define WS_ATT  (WS_VT  + (size_t)B_*H_*SK_*HD_*2)   // attn out f16        [4096 x 2048]

// ---- d_out layout (floats) ----
#define OUT_K ((size_t)M_*D_)                         // new_k [B,H,SK,HD]
#define OUT_V (OUT_K + (size_t)B_*H_*SK_*HD_)         // new_v [B,H,SK,HD]

union U16x8 { v16h h; unsigned int u[8]; uint4 q[2]; };

// ln(10000)/64
#define ROPE_C 0.14391156505f

// ---------------- conversion kernel (vectorized f32 -> f16) ----------------
__global__ void k_cvt_f32_f16(const float* __restrict__ src, _Float16* __restrict__ dst, int n4) {
    int i = blockIdx.x*256 + threadIdx.x;
    if (i < n4) {
        float4 v = ((const float4*)src)[i];
        union { _Float16 h[4]; uint2 u; } t;
        t.h[0] = (_Float16)v.x; t.h[1] = (_Float16)v.y;
        t.h[2] = (_Float16)v.z; t.h[3] = (_Float16)v.w;
        ((uint2*)dst)[i] = t.u;
    }
}

// past cache -> f16 caches + new_k/new_v past halves
__global__ void k_copy_past(const float* __restrict__ pk, const float* __restrict__ pv,
                            _Float16* __restrict__ kvis, _Float16* __restrict__ vt,
                            float* __restrict__ outk, float* __restrict__ outv) {
    int i = blockIdx.x*256 + threadIdx.x;   // B*H*256*64
    int d  = i & 63;
    int s  = (i >> 6) & 255;
    int bh = i >> 14;
    size_t pi = ((size_t)bh*512 + 256 + s)*64 + d;
    float kvv = pk[pi], vvv = pv[pi];
    kvis[((size_t)bh*512 + s)*64 + d] = (_Float16)kvv;
    vt  [((size_t)bh*64 + d)*512 + s] = (_Float16)vvv;
    outk[((size_t)bh*512 + s)*64 + d] = kvv;
    outv[((size_t)bh*512 + s)*64 + d] = vvv;
}

// ---------------- tiled WMMA GEMM, C[m,n] = sum_k A[m,k]*W[n,k] ----------------
// Block tile 128x256, 8 waves (2x4), 64x64 per wave (16 wmma / k-step).
// mode 1=Q(rope,scale->qrot) 2=K(rope->kvis, raw->new_k) 3=V(->vt, raw->new_v) 4=final(->d_out f32)
__global__ __launch_bounds__(256) void k_gemm(const _Float16* __restrict__ A,
                                              const _Float16* __restrict__ Bw,  // f16 W, row-major [n][k]
                                              _Float16* __restrict__ outH,
                                              float* __restrict__ outF,
                                              const int* __restrict__ posp,
                                              int mode)
{
    __shared__ _Float16 lA[128][40];   // [m][k]
    __shared__ _Float16 lB[256][40];   // [n][k]  (K-pairs contiguous for B fragments)
    int tid  = threadIdx.x;
    int wave = tid >> 5, lane = tid & 31;
    int half = lane >> 4, l16 = lane & 15;
    int wm = wave >> 2, wn = wave & 3;          // 2 x 4 wave grid, each wave 64x64
    int mblk = blockIdx.y * 128, nblk = blockIdx.x * 256;

    v8f acc[4][4];
    #pragma unroll
    for (int a = 0; a < 4; a++)
        #pragma unroll
        for (int b = 0; b < 4; b++)
            #pragma unroll
            for (int r = 0; r < 8; r++) acc[a][b][r] = 0.f;

    int arow = tid >> 1, ac = (tid & 1) * 16;
    for (int k0 = 0; k0 < 2048; k0 += 32) {
        { // A tile 128x32
            const uint4* s = (const uint4*)(A + (size_t)(mblk + arow)*2048 + k0 + ac);
            uint4* d = (uint4*)&lA[arow][ac];
            d[0] = s[0]; d[1] = s[1];
        }
        #pragma unroll
        for (int rr = 0; rr < 2; rr++) { // B tile 256x32
            int row = rr*128 + arow;
            const uint4* s = (const uint4*)(Bw + (size_t)(nblk + row)*2048 + k0 + ac);
            uint4* d = (uint4*)&lB[row][ac];
            d[0] = s[0]; d[1] = s[1];
        }
        __syncthreads();

        U16x8 af[4], bf[4];
        #pragma unroll
        for (int sm = 0; sm < 4; sm++) {      // A frag: v0-3 = K 0..7(+8h), v4-7 = K 16..23(+8h)
            int mr = wm*64 + sm*16 + l16;
            af[sm].q[0] = *(const uint4*)&lA[mr][8*half];
            af[sm].q[1] = *(const uint4*)&lA[mr][16 + 8*half];
        }
        #pragma unroll
        for (int sn = 0; sn < 4; sn++) {      // B frag: v0-3 = K 0..7(+8h), v4-7 = K 16..23(+8h)
            int nc = wn*64 + sn*16 + l16;
            bf[sn].q[0] = *(const uint4*)&lB[nc][8*half];
            bf[sn].q[1] = *(const uint4*)&lB[nc][16 + 8*half];
        }
        #pragma unroll
        for (int sm = 0; sm < 4; sm++)
            #pragma unroll
            for (int sn = 0; sn < 4; sn++)
                acc[sm][sn] = __builtin_amdgcn_wmma_f32_16x16x32_f16(
                    false, af[sm].h, false, bf[sn].h, (short)0, acc[sm][sn], false, false);
        __syncthreads();
    }

    int pos0 = (mode == 1 || mode == 2) ? *posp : 0;
    #pragma unroll
    for (int sm = 0; sm < 4; sm++)
      #pragma unroll
      for (int sn = 0; sn < 4; sn++)
        #pragma unroll
        for (int r = 0; r < 8; r++) {
            int m = mblk + wm*64 + sm*16 + r + 8*half;
            int n = nblk + wn*64 + sn*16 + l16;
            float val = acc[sm][sn][r];
            if (mode == 4) {
                outF[(size_t)m*2048 + n] = val;
            } else {
                int b = m >> 8, t = m & 255;
                int h = n >> 6, d = n & 63;
                int bh = b*32 + h;
                if (mode == 3) {
                    outH[((size_t)bh*64 + d)*512 + 256 + t] = (_Float16)val;
                    outF[((size_t)bh*512 + 256 + t)*64 + d] = val;
                } else {
                    // RoPE: adjacent d-columns live in adjacent lanes
                    float partner = __shfl_xor(val, 1);
                    float fr  = __expf(-(float)(d & ~1) * ROPE_C);  // 10000^(-(2p)/64)
                    float ang = (float)(pos0 + t) * fr;
                    float cs = __cosf(ang), si = __sinf(ang);
                    float rv = val*cs + partner * ((d & 1) ? si : -si);
                    if (mode == 1) {
                        outH[((size_t)bh*256 + t)*64 + d] = (_Float16)(rv * 0.125f); // * Hd^-0.5
                    } else {
                        outH[((size_t)bh*512 + 256 + t)*64 + d] = (_Float16)rv;
                        outF[((size_t)bh*512 + 256 + t)*64 + d] = val;  // pre-rope for new_k
                    }
                }
            }
        }
}

// ---------------- flash attention: one wave per 16-row q tile ----------------
__global__ __launch_bounds__(256) void k_attn(const _Float16* __restrict__ qrot,
                                              const _Float16* __restrict__ kvis,
                                              const _Float16* __restrict__ vt,
                                              _Float16* __restrict__ attO)
{
    __shared__ _Float16 stage[8][16*32];     // per-wave zero-padded 16x32 P tile
    int bh   = blockIdx.x >> 1, g = blockIdx.x & 1;
    int wave = threadIdx.x >> 5, lane = threadIdx.x & 31;
    int half = lane >> 4, l16 = lane & 15;
    int qt   = g*8 + wave;

    { // zero staging; cols 16..31 stay zero forever
        unsigned int* p = (unsigned int*)&stage[wave][0];
        #pragma unroll
        for (int j = 0; j < 8; j++) p[lane + j*32] = 0u;
    }

    U16x8 aq0, aq1;
    {
        size_t base = ((size_t)bh*256 + qt*16 + l16)*64;
        aq0.q[0] = *(const uint4*)&qrot[base + 8*half];
        aq0.q[1] = *(const uint4*)&qrot[base + 16 + 8*half];
        aq1.q[0] = *(const uint4*)&qrot[base + 32 + 8*half];
        aq1.q[1] = *(const uint4*)&qrot[base + 48 + 8*half];
    }

    float m8[8], l8[8];
    #pragma unroll
    for (int r = 0; r < 8; r++) { m8[r] = -__builtin_inff(); l8[r] = 0.f; }
    v8f O[4];
    #pragma unroll
    for (int c = 0; c < 4; c++)
        #pragma unroll
        for (int r = 0; r < 8; r++) O[c][r] = 0.f;

    int ktmax = qt + 16;                      // band: kk <= qi + 256
    for (int kt = 0; kt <= ktmax; kt++) {
        U16x8 bk0, bk1;                       // B frag of K: K-dim = d, pairs contiguous in kvis rows
        size_t kbase = ((size_t)bh*512 + kt*16 + l16)*64;
        bk0.q[0] = *(const uint4*)&kvis[kbase + 8*half];
        bk0.q[1] = *(const uint4*)&kvis[kbase + 16 + 8*half];
        bk1.q[0] = *(const uint4*)&kvis[kbase + 32 + 8*half];
        bk1.q[1] = *(const uint4*)&kvis[kbase + 48 + 8*half];

        v8f S;
        #pragma unroll
        for (int r = 0; r < 8; r++) S[r] = 0.f;
        S = __builtin_amdgcn_wmma_f32_16x16x32_f16(false, aq0.h, false, bk0.h, (short)0, S, false, false);
        S = __builtin_amdgcn_wmma_f32_16x16x32_f16(false, aq1.h, false, bk1.h, (short)0, S, false, false);

        if (kt == ktmax) {                    // diagonal tile: mask i < j
            #pragma unroll
            for (int r = 0; r < 8; r++)
                if ((r + 8*half) < l16) S[r] = -__builtin_inff();
        }

        #pragma unroll
        for (int r = 0; r < 8; r++) {
            float rm = S[r];
            #pragma unroll
            for (int off = 1; off < 16; off <<= 1) rm = fmaxf(rm, __shfl_xor(rm, off));
            float nm = fmaxf(m8[r], rm);
            float sc = __expf(m8[r] - nm);
            m8[r] = nm;
            float pv = __expf(S[r] - nm);
            float rs = pv;
            #pragma unroll
            for (int off = 1; off < 16; off <<= 1) rs += __shfl_xor(rs, off);
            l8[r] = l8[r]*sc + rs;
            #pragma unroll
            for (int c = 0; c < 4; c++) O[c][r] *= sc;
            stage[wave][(r + 8*half)*32 + l16] = (_Float16)pv;
        }
        asm volatile("s_wait_dscnt 0" ::: "memory");

        U16x8 ap;                             // P as A fragment from LDS (cols 16..31 zero)
        ap.q[0] = *(const uint4*)&stage[wave][l16*32 + 16*half];
        ap.q[1] = *(const uint4*)&stage[wave][l16*32 + 16*half + 8];

        #pragma unroll
        for (int c = 0; c < 4; c++) {
            U16x8 bv;                         // B frag of V: K-dim = s, contiguous in vt rows
            int d = c*16 + l16;
            int s0 = kt*16 + 16*half;
            if (s0 > 496) s0 = 496;           // keep in bounds; garbage * zero-P
            const _Float16* vb = &vt[((size_t)bh*64 + d)*512 + s0];
            bv.q[0] = *(const uint4*)vb;
            bv.q[1] = *(const uint4*)(vb + 8);
            O[c] = __builtin_amdgcn_wmma_f32_16x16x32_f16(false, ap.h, false, bv.h, (short)0, O[c], false, false);
        }
    }

    int b = bh >> 5, h = bh & 31;
    #pragma unroll
    for (int r = 0; r < 8; r++) {
        float inv = 1.0f / l8[r];
        int t = qt*16 + r + 8*half;
        #pragma unroll
        for (int c = 0; c < 4; c++)
            attO[((size_t)b*256 + t)*2048 + h*64 + c*16 + l16] = (_Float16)(O[c][r] * inv);
    }
}

extern "C" void kernel_launch(void* const* d_in, const int* in_sizes, int n_in,
                              void* d_out, int out_size, void* d_ws, size_t ws_size,
                              hipStream_t stream) {
    const float* x   = (const float*)d_in[0];
    const float* pk  = (const float*)d_in[1];
    const float* pv  = (const float*)d_in[2];
    const float* Wq  = (const float*)d_in[3];
    const float* Wk  = (const float*)d_in[4];
    const float* Wv  = (const float*)d_in[5];
    const float* Wo  = (const float*)d_in[6];
    const int*  posp = (const int*)d_in[7];
    float* out = (float*)d_out;
    char*  ws  = (char*)d_ws;

    _Float16* xa   = (_Float16*)(ws + WS_XA);
    _Float16* wq16 = (_Float16*)(ws + WS_W16);
    _Float16* wk16 = wq16 + (size_t)D_*D_;
    _Float16* wv16 = wk16 + (size_t)D_*D_;
    _Float16* wo16 = wv16 + (size_t)D_*D_;
    _Float16* qrot = (_Float16*)(ws + WS_QROT);
    _Float16* kvis = (_Float16*)(ws + WS_KVIS);
    _Float16* vt   = (_Float16*)(ws + WS_VT);
    _Float16* attO = (_Float16*)(ws + WS_ATT);

    k_cvt_f32_f16<<<(M_*D_/4)/256, 256, 0, stream>>>(x,  xa,   M_*D_/4);
    k_cvt_f32_f16<<<(D_*D_/4)/256, 256, 0, stream>>>(Wq, wq16, D_*D_/4);
    k_cvt_f32_f16<<<(D_*D_/4)/256, 256, 0, stream>>>(Wk, wk16, D_*D_/4);
    k_cvt_f32_f16<<<(D_*D_/4)/256, 256, 0, stream>>>(Wv, wv16, D_*D_/4);
    k_cvt_f32_f16<<<(D_*D_/4)/256, 256, 0, stream>>>(Wo, wo16, D_*D_/4);

    dim3 gg(D_/256, M_/128);
    k_gemm<<<gg, 256, 0, stream>>>(xa, wq16, qrot, nullptr,     posp, 1);
    k_gemm<<<gg, 256, 0, stream>>>(xa, wk16, kvis, out + OUT_K, posp, 2);
    k_gemm<<<gg, 256, 0, stream>>>(xa, wv16, vt,   out + OUT_V, posp, 3);

    k_copy_past<<<(B_*H_*256*64)/256, 256, 0, stream>>>(pk, pv, kvis, vt, out + OUT_K, out + OUT_V);

    k_attn<<<B_*H_*2, 256, 0, stream>>>(qrot, kvis, vt, attO);

    k_gemm<<<gg, 256, 0, stream>>>(attO, wo16, nullptr, out, posp, 4);
}
